// LocalGNN_DB_6777458393746
// MI455X (gfx1250) — matrix operand order repaired and checked by reading, not compiled, not run on hardware
//
#include <hip/hip_runtime.h>
#include <hip/hip_bf16.h>
#include <math.h>

typedef __bf16 bf16;
typedef __attribute__((ext_vector_type(16))) __bf16 v16bf;
typedef __attribute__((ext_vector_type(8)))  float  v8f;

#define N_NODES   256
#define B_DIM     8
#define T_DIM     64
#define BT_TOTAL  (B_DIM * T_DIM)
#define F0_DIM    12
#define F1_DIM    64
#define F2_DIM    32
#define THREADS   512
#define BT_STR    40     // K-transposed S-panel LDS row stride (bf16): 80B rows,
                         // 16B-aligned (=> ds_load_b128 frags), 20dw stride is
                         // bank-conflict-free for 16 consecutive n (20n mod 64 distinct)
#define A_STR     264    // A-operand LDS row stride (bf16): 528B rows, 16B aligned

__device__ __forceinline__ v8f zero_v8f() {
  v8f v;
#pragma unroll
  for (int i = 0; i < 8; ++i) v[i] = 0.0f;
  return v;
}

__device__ __forceinline__ v8f wmma_bf16(v16bf a, v16bf b, v8f c) {
  // D = A(16x32 bf16) * B(32x16 bf16) + C(16x16 f32)
  return __builtin_amdgcn_wmma_f32_16x16x32_bf16(false, a, false, b, (short)0, c,
                                                 false, false);
}

// Pack two f32 -> packed bf16x2 and store as one 4B-aligned ds_store_b32.
__device__ __forceinline__ void pack_store2(bf16* dst, float lo, float hi) {
  union { bf16 h[2]; unsigned u; } pk;
  pk.h[0] = (bf16)lo;   // k (even)
  pk.h[1] = (bf16)hi;   // k+1
  *(unsigned*)dst = pk.u;  // lowers to v_cvt_pk_bf16_f32 + ds_store_b32
}

// A fragment (16x32 bf16): lanes 0-15 M=lane, K=0..15 contiguous;
// lanes 16-31 same M rows with K=16..31 (matches B fragment K order).
// Rows are 16B aligned -> lowers to 2x ds_load_b128.
__device__ __forceinline__ v16bf load_a_frag(const bf16* Al, int m0, int kk, int lane) {
  const bf16* row = Al + (m0 + (lane & 15)) * A_STR;
  const int kb = (lane & 16) ? 16 : 0;
  v16bf a;
#pragma unroll
  for (int e = 0; e < 8; ++e) {
    const int k = kk + kb + 2 * e;  // pairs (k,k+1)
    a[2 * e]     = row[k];
    a[2 * e + 1] = row[k + 1];
  }
  return a;
}

// B fragment (32x16 bf16): lane n = lane&15; lanes 0-15 hold K=0..15,
// lanes 16-31 hold K=16..31. Panel is K-transposed in LDS so each lane's 16
// values are contiguous and 16B aligned -> 2x ds_load_b128.
__device__ __forceinline__ v16bf load_b_frag(const bf16* Bt, int n0, int lane) {
  const bf16* col = Bt + (n0 + (lane & 15)) * BT_STR + ((lane & 16) ? 16 : 0);
  v16bf b;
#pragma unroll
  for (int e = 0; e < 16; ++e) b[e] = col[e];
  return b;
}

// Stage the A operand (gsrc x 256 bf16, zero-padded to MR rows; null => zeros).
template <int MR>
__device__ __forceinline__ void stage_a(bf16* Al, const bf16* Asrc, int gsrc, int tid) {
  for (int ch = tid; ch < MR * 32; ch += THREADS) {
    const int g = ch >> 5;
    const int c = (ch & 31) * 8;
    uint4 v;
    v.x = v.y = v.z = v.w = 0u;
    if (Asrc != nullptr && g < gsrc) v = *(const uint4*)(Asrc + (size_t)g * N_NODES + c);
    *(uint4*)(Al + g * A_STR + c) = v;  // 16B-aligned ds_store_b128
  }
}

// Full K=256 GEMM: 8 K-steps of 32. The fp32 S panel for step kk+1 is loaded
// into registers while the WMMAs of step kk execute (software pipeline), so the
// global-load latency is off the barrier-to-barrier critical path. LDS panel is
// single-buffered; conversion fp32->bf16 happens on the register->LDS store.
// Each thread stages a row pair (2p, 2p+1) x 8 columns: the pair packs to one
// v_cvt_pk_bf16_f32 + one conflict-free ds_store_b32 per column.
template <int NPW>
__device__ __forceinline__ void gemm_run(const bf16* Al, bf16* Bt, const float* Sp,
                                         v8f* acc, int mt, int nq, int tid, int lane) {
  const int p  = tid & 15;        // row pair: rows 2p and 2p+1 of the 32-row panel
  const int c0 = (tid >> 4) * 8;  // columns c0..c0+7
  float4 lo0, lo1, hi0, hi1;      // lo = row 2p, hi = row 2p+1
  auto load_panel = [&](int kk) {
    const float* r0 = Sp + (size_t)(kk + 2 * p) * N_NODES + c0;
    const float* r1 = r0 + N_NODES;
    lo0 = *(const float4*)(r0);
    lo1 = *(const float4*)(r0 + 4);
    hi0 = *(const float4*)(r1);
    hi1 = *(const float4*)(r1 + 4);
  };
  auto store_panel = [&]() {
    bf16* d = Bt + c0 * BT_STR + 2 * p;
    pack_store2(d + 0 * BT_STR, lo0.x, hi0.x);
    pack_store2(d + 1 * BT_STR, lo0.y, hi0.y);
    pack_store2(d + 2 * BT_STR, lo0.z, hi0.z);
    pack_store2(d + 3 * BT_STR, lo0.w, hi0.w);
    pack_store2(d + 4 * BT_STR, lo1.x, hi1.x);
    pack_store2(d + 5 * BT_STR, lo1.y, hi1.y);
    pack_store2(d + 6 * BT_STR, lo1.z, hi1.z);
    pack_store2(d + 7 * BT_STR, lo1.w, hi1.w);
  };
  load_panel(0);
  for (int kk = 0; kk < N_NODES; kk += 32) {
    __syncthreads();                 // previous panel fully consumed (+A staged)
    store_panel();                   // cvt_pk_bf16 + ds_store_b32 (K-transposed)
    __syncthreads();                 // panel visible to all waves
    if (kk + 32 < N_NODES) {
      load_panel(kk + 32);           // overlaps with fragment loads + WMMAs below
      if (kk + 64 < N_NODES)         // warm L2/L0 two panels ahead (no counters)
        __builtin_prefetch(Sp + (size_t)(kk + 64 + 2 * p) * N_NODES + c0, 0, 1);
    }
    const v16bf a = load_a_frag(Al, mt * 16, kk, lane);
#pragma unroll
    for (int j = 0; j < NPW; ++j) {
      const v16bf b = load_b_frag(Bt, (nq * NPW + j) * 16, lane);
      acc[j] = wmma_bf16(a, b, acc[j]);
    }
  }
}

// Z[b,t] = A[b,t-1] @ S[b,t]  (one workgroup per (b,t); output bf16)
template <int MR, int GSRC, int GOUT>
__global__ __launch_bounds__(THREADS) void diffuse_kernel(
    const bf16* __restrict__ Aall, const float* __restrict__ Sall,
    bf16* __restrict__ Zall) {
  constexpr int MT = MR / 16;  // M tiles
  constexpr int NQ = 16 / MT;  // waves along N
  constexpr int NPW = MT;      // 16 N-tiles / NQ
  __shared__ alignas(16) bf16 Al[MR * A_STR];
  __shared__ alignas(16) bf16 Bt[N_NODES * BT_STR];
  const int bt = blockIdx.x;
  const int t = bt & (T_DIM - 1);
  const int tid = threadIdx.x, wid = tid >> 5, lane = tid & 31;
  const bf16* Asrc = (t > 0) ? (Aall + (size_t)(bt - 1) * (GSRC * N_NODES)) : nullptr;
  const float* Sp = Sall + (size_t)bt * (N_NODES * N_NODES);
  stage_a<MR>(Al, Asrc, GSRC, tid);
  v8f acc[NPW];
#pragma unroll
  for (int j = 0; j < NPW; ++j) acc[j] = zero_v8f();
  const int mt = wid / NQ, nq = wid % NQ;
  gemm_run<NPW>(Al, Bt, Sp, acc, mt, nq, tid, lane);
  bf16* Zo = Zall + (size_t)bt * (GOUT * N_NODES);
  const int nc = lane & 15;
  const int mr = mt * 16 + ((lane & 16) ? 8 : 0);  // C layout: VGPR r -> M = r (+8)
#pragma unroll
  for (int j = 0; j < NPW; ++j) {
    const int n = (nq * NPW + j) * 16 + nc;
#pragma unroll
    for (int r = 0; r < 8; ++r) {
      const int m = mr + r;
      if (m < GOUT) Zo[m * N_NODES + n] = (bf16)acc[j][r];
    }
  }
}

// Z12 = Z11[b,t-1] @ S[b,t] (WMMA), then Y1 = tanh(W1k0@X + W1k1@Z11 + W1k2@Z12 + b1)
__global__ __launch_bounds__(THREADS) void layer1_kernel(
    const bf16* __restrict__ Xall, const bf16* __restrict__ Z11all,
    const float* __restrict__ Sall, const float* __restrict__ W1,
    const float* __restrict__ b1, bf16* __restrict__ Y1all) {
  __shared__ alignas(16) bf16 Al[16 * A_STR];
  __shared__ alignas(16) bf16 Bt[N_NODES * BT_STR];
  __shared__ alignas(16) float Z12l[16 * N_NODES];
  const int bt = blockIdx.x;
  const int t = bt & (T_DIM - 1);
  const int tid = threadIdx.x, wid = tid >> 5, lane = tid & 31;
  const bf16* Asrc = (t > 0) ? (Z11all + (size_t)(bt - 1) * (F0_DIM * N_NODES)) : nullptr;
  const float* Sp = Sall + (size_t)bt * (N_NODES * N_NODES);
  stage_a<16>(Al, Asrc, F0_DIM, tid);
  v8f acc = zero_v8f();
  gemm_run<1>(Al, Bt, Sp, &acc, 0, wid, tid, lane);
  __syncthreads();
  {
    const int n = wid * 16 + (lane & 15);
    const int mr = (lane & 16) ? 8 : 0;
#pragma unroll
    for (int r = 0; r < 8; ++r) Z12l[(mr + r) * N_NODES + n] = acc[r];
  }
  __syncthreads();
  const bf16* Xt = Xall + (size_t)bt * (F0_DIM * N_NODES);
  const bf16* Z11t = Z11all + (size_t)bt * (F0_DIM * N_NODES);
  bf16* Yo = Y1all + (size_t)bt * (F1_DIM * N_NODES);
  for (int o = tid; o < F1_DIM * N_NODES; o += THREADS) {
    const int n = o & (N_NODES - 1);
    const int f = o >> 8;                       // uniform across a wave -> s_loads for W1
    const float* w = W1 + f * (3 * F0_DIM);     // W1[f,0,k,g]
    float s = b1[f];
#pragma unroll
    for (int g = 0; g < F0_DIM; ++g) {
      s += w[g]              * (float)Xt[g * N_NODES + n];
      s += w[F0_DIM + g]     * (float)Z11t[g * N_NODES + n];
      s += w[2 * F0_DIM + g] * Z12l[g * N_NODES + n];
    }
    Yo[f * N_NODES + n] = (bf16)tanhf(s);
  }
}

// Z22 = Z21[b,t-1] @ S[b,t] (WMMA), Y2 = tanh(W2-combine), then per-node MLP readout.
__global__ __launch_bounds__(THREADS) void layer2_kernel(
    const bf16* __restrict__ Y1all, const bf16* __restrict__ Z21all,
    const float* __restrict__ Sall, const float* __restrict__ W2,
    const float* __restrict__ b2, const float* __restrict__ A1,
    const float* __restrict__ c1, const float* __restrict__ A2,
    const float* __restrict__ c2, float* __restrict__ out) {
  // 54272B pool: GEMM stage uses Al(33792)+Bt(20480); combine stage reuses it
  // as Z22 bf16 (32768) + Y2 bf16 (16384).
  __shared__ alignas(16) char smem[64 * A_STR * 2 + N_NODES * BT_STR * 2];
  bf16* Al = (bf16*)smem;
  bf16* Bt = (bf16*)(smem + 64 * A_STR * 2);
  const int bt = blockIdx.x;
  const int t = bt & (T_DIM - 1);
  const int tid = threadIdx.x, wid = tid >> 5, lane = tid & 31;
  const bf16* Asrc = (t > 0) ? (Z21all + (size_t)(bt - 1) * (F1_DIM * N_NODES)) : nullptr;
  const float* Sp = Sall + (size_t)bt * (N_NODES * N_NODES);
  stage_a<64>(Al, Asrc, F1_DIM, tid);
  v8f acc[4];
#pragma unroll
  for (int j = 0; j < 4; ++j) acc[j] = zero_v8f();
  const int mt = wid >> 2, nq = wid & 3;
  gemm_run<4>(Al, Bt, Sp, acc, mt, nq, tid, lane);
  __syncthreads();  // all waves done with Al/Bt; safe to alias
  bf16* Z22l = (bf16*)smem;
  bf16* Y2l = (bf16*)(smem + 64 * N_NODES * 2);
  {
    const int nc = lane & 15;
    const int mr = mt * 16 + ((lane & 16) ? 8 : 0);
#pragma unroll
    for (int j = 0; j < 4; ++j) {
      const int n = (nq * 4 + j) * 16 + nc;
#pragma unroll
      for (int r = 0; r < 8; ++r) Z22l[(mr + r) * N_NODES + n] = (bf16)acc[j][r];
    }
  }
  __syncthreads();
  const bf16* Y1t = Y1all + (size_t)bt * (F1_DIM * N_NODES);
  const bf16* Z21t = Z21all + (size_t)bt * (F1_DIM * N_NODES);
  for (int o = tid; o < F2_DIM * N_NODES; o += THREADS) {
    const int n = o & (N_NODES - 1);
    const int f = o >> 8;
    const float* w = W2 + f * (3 * F1_DIM);  // W2[f,0,k,g]
    float s = b2[f];
#pragma unroll 4
    for (int g = 0; g < F1_DIM; ++g) {
      s += w[g]              * (float)Y1t[g * N_NODES + n];
      s += w[F1_DIM + g]     * (float)Z21t[g * N_NODES + n];
      s += w[2 * F1_DIM + g] * (float)Z22l[g * N_NODES + n];
    }
    Y2l[f * N_NODES + n] = (bf16)tanhf(s);
  }
  __syncthreads();
  if (tid < N_NODES) {  // one thread per node: 32->32 tanh -> 2
    float y[F2_DIM];
#pragma unroll
    for (int f = 0; f < F2_DIM; ++f) y[f] = (float)Y2l[f * N_NODES + tid];
    float u[32];
    for (int r = 0; r < 32; ++r) {
      float s = c1[r];
      const float* a1 = A1 + r * F2_DIM;
#pragma unroll
      for (int f = 0; f < F2_DIM; ++f) s += a1[f] * y[f];
      u[r] = tanhf(s);
    }
    float* op = out + (size_t)bt * (2 * N_NODES);
#pragma unroll
    for (int q = 0; q < 2; ++q) {
      float s = c2[q];
      const float* a2 = A2 + q * 32;
#pragma unroll
      for (int f = 0; f < 32; ++f) s += a2[f] * u[f];
      op[q * N_NODES + tid] = s;
    }
  }
}

// fp32 -> bf16, 8 elements/thread (two float4 loads, one uint4 store)
__global__ __launch_bounds__(256) void cvt_kernel(const float* __restrict__ src,
                                                  bf16* __restrict__ dst, int n8) {
  const int i = blockIdx.x * 256 + threadIdx.x;
  if (i >= n8) return;
  const float4* s = (const float4*)src + (size_t)i * 2;
  const float4 v0 = s[0], v1 = s[1];
  union { bf16 h[8]; uint4 u; } o;
  o.h[0] = (bf16)v0.x; o.h[1] = (bf16)v0.y; o.h[2] = (bf16)v0.z; o.h[3] = (bf16)v0.w;
  o.h[4] = (bf16)v1.x; o.h[5] = (bf16)v1.y; o.h[6] = (bf16)v1.z; o.h[7] = (bf16)v1.w;
  ((uint4*)dst)[i] = o.u;
}

extern "C" void kernel_launch(void* const* d_in, const int* in_sizes, int n_in,
                              void* d_out, int out_size, void* d_ws, size_t ws_size,
                              hipStream_t stream) {
  (void)in_sizes; (void)n_in; (void)out_size; (void)ws_size;
  const float* x  = (const float*)d_in[0];   // (8,64,12,256)
  const float* S  = (const float*)d_in[1];   // (8,64,1,256,256)
  const float* W1 = (const float*)d_in[2];   // (64,1,3,12)
  const float* b1 = (const float*)d_in[3];   // (64,1)
  const float* W2 = (const float*)d_in[4];   // (32,1,3,64)
  const float* b2 = (const float*)d_in[5];   // (32,1)
  const float* A1 = (const float*)d_in[6];   // (32,32)
  const float* c1 = (const float*)d_in[7];   // (32,)
  const float* A2 = (const float*)d_in[8];   // (2,32)
  const float* c2 = (const float*)d_in[9];   // (2,)
  float* out = (float*)d_out;                // (8,64,2,256)

  // Workspace layout (bf16 intermediates, ~38 MB total)
  bf16* Xbf = (bf16*)d_ws;
  bf16* Z11 = Xbf + (size_t)BT_TOTAL * F0_DIM * N_NODES;
  bf16* Y1  = Z11 + (size_t)BT_TOTAL * F0_DIM * N_NODES;
  bf16* Z21 = Y1  + (size_t)BT_TOTAL * F1_DIM * N_NODES;

  const int xn8 = (BT_TOTAL * F0_DIM * N_NODES) / 8;  // 196608
  cvt_kernel<<<(xn8 + 255) / 256, 256, 0, stream>>>(x, Xbf, xn8);
  diffuse_kernel<16, F0_DIM, F0_DIM><<<BT_TOTAL, THREADS, 0, stream>>>(Xbf, S, Z11);
  layer1_kernel<<<BT_TOTAL, THREADS, 0, stream>>>(Xbf, Z11, S, W1, b1, Y1);
  diffuse_kernel<64, F1_DIM, F1_DIM><<<BT_TOTAL, THREADS, 0, stream>>>(Y1, S, Z21);
  layer2_kernel<<<BT_TOTAL, THREADS, 0, stream>>>(Y1, Z21, S, W2, b2, A1, c1, A2, c2,
                                                  out);
}
